// BuNNLayer_5875515261229
// MI455X (gfx1250) — matrix operand order repaired and checked by reading, not compiled, not run on hardware
//
#include <hip/hip_runtime.h>
#include <cmath>

#define C_DIM 512      // channels
#define B_DIM 128      // bundles (= C/4), also float4 columns per row

typedef __attribute__((ext_vector_type(16))) __bf16 v16bf;
typedef __attribute__((ext_vector_type(8)))  float  v8f;

__device__ __forceinline__ unsigned short f32_to_bf16(float f) {
  union { float f; unsigned u; } v; v.f = f;
  unsigned u = v.u;
  u += 0x7FFFu + ((u >> 16) & 1u);          // round-to-nearest-even
  return (unsigned short)(u >> 16);
}

// ---------------------------------------------------------------------------
// CDNA5 async Global->LDS copy (ASYNCcnt-tracked, bypasses VGPRs).
// Low 32 bits of a generic pointer to LDS are the LDS byte offset (flat
// aperture mapping, ISA 10.2), so truncation yields the DS address.
// ---------------------------------------------------------------------------
__device__ __forceinline__ void async_copy_b128(void* lds_ptr, const void* gptr) {
  unsigned lds = (unsigned)(unsigned long long)lds_ptr;
  asm volatile("global_load_async_to_lds_b128 %0, %1, off"
               :: "v"(lds), "v"(gptr) : "memory");
}

__device__ __forceinline__ void wait_async0() {
#if __has_builtin(__builtin_amdgcn_s_wait_asynccnt)
  __builtin_amdgcn_s_wait_asynccnt(0);
#else
  asm volatile("s_wait_asynccnt 0x0" ::: "memory");
#endif
}

// ---------------------------------------------------------------------------
// degree / edge weights
// ---------------------------------------------------------------------------
__global__ void deg_kernel(const long long* __restrict__ src, float* __restrict__ deg, int E) {
  int e = blockIdx.x * 256 + threadIdx.x;
  if (e < E) unsafeAtomicAdd(&deg[(int)src[e]], 1.0f);
}

__global__ void ew_kernel(const long long* __restrict__ dst, const float* __restrict__ deg,
                          float* __restrict__ ew, int E) {
  int e = blockIdx.x * 256 + threadIdx.x;
  if (e < E) ew[e] = 1.0f / deg[(int)dst[e]];
}

// ---------------------------------------------------------------------------
// f32 -> bf16 conversion
// ---------------------------------------------------------------------------
__global__ void cvt_bf16_k(const float* __restrict__ in, unsigned short* __restrict__ out, int n) {
  int i = blockIdx.x * 256 + threadIdx.x;
  if (i < n) out[i] = f32_to_bf16(in[i]);
}

// ---------------------------------------------------------------------------
// WMMA bf16 GEMM: C[M,Nc] = A[M,K] * B[K,Nc] (+bias) with fused epilogues
//   EPI 0: f32 out (+bias)
//   EPI 1: gelu(+bias) -> bf16 out
//   EPI 2: theta=+bias -> cos/sin f32 outs
// Block tile 128x64, 8 waves (4x2), each wave 32x32 via 2x2 v_wmma 16x16x32.
// A tile staged via global_load_async_to_lds_b128, double-buffered LDS so the
// next chunk's async copies overlap the current chunk's WMMAs.
// ---------------------------------------------------------------------------
template <int EPI>
__global__ __launch_bounds__(256) void gemm_bf16_wmma(
    const unsigned short* __restrict__ A,
    const unsigned short* __restrict__ Bw,
    const float* __restrict__ bias,
    float* __restrict__ outF,
    unsigned short* __restrict__ outB,
    float* __restrict__ outC,
    float* __restrict__ outS,
    int M, int Nc, int K)
{
  // 128 x 32 (+8 pad): row stride 80B (multiple of 16B for b128 access)
  __shared__ __align__(16) unsigned short As[2][128][40];
  // n-major: 64 cols x 32 k (+8 pad)
  __shared__ __align__(16) unsigned short Bs[2][64][40];

  const int tid   = threadIdx.x;
  const int lane  = tid & 31;
  const int wave  = tid >> 5;
  const int lm    = lane & 15;
  const int half  = lane >> 4;
  const int waveM = wave >> 1;   // 0..3
  const int waveN = wave & 1;    // 0..1
  const int mBlk  = blockIdx.y * 128;
  const int nBlk  = blockIdx.x * 64;

  v8f acc[2][2] = {};

  // stage one K-chunk (32 wide) into LDS buffer `buf`
  auto stage = [&](int k0, int buf) {
    // A tile 128x32: 2 async b128 copies per thread (rows >= M left stale;
    // their accumulator rows are never stored)
    #pragma unroll
    for (int it = 0; it < 2; ++it) {
      int idx  = tid + it * 256;
      int row  = idx >> 2;
      int colc = (idx & 3) * 8;
      int gr   = mBlk + row;
      if (gr < M)
        async_copy_b128(&As[buf][row][colc], A + (size_t)gr * K + k0 + colc);
    }
    // B tile 32x64, transposed to n-major via register path
    int k  = tid >> 3;
    int nc = (tid & 7) * 8;
    uint4 val = *(const uint4*)(Bw + (size_t)(k0 + k) * Nc + nBlk + nc);
    union { uint4 q; unsigned short s[8]; } u; u.q = val;
    #pragma unroll
    for (int j = 0; j < 8; ++j) Bs[buf][nc + j][k] = u.s[j];
  };

  stage(0, 0);
  wait_async0();         // __syncthreads does NOT flush ASYNCcnt
  __syncthreads();

  int buf = 0;
  for (int k0 = 0; k0 < K; k0 += 32, buf ^= 1) {
    // ---- build fragments per ISA 7.12.2 16-bit A/B layouts ----
    v16bf afr[2], bfr[2];
    #pragma unroll
    for (int i = 0; i < 2; ++i) {
      union { unsigned u[8]; v16bf v; } f;
      const unsigned short* row = As[buf][waveM * 32 + i * 16 + lm];
      int b0 = 8 * half;
      #pragma unroll
      for (int r = 0; r < 4; ++r) {                   // K(j)=16*(j/8)+8*half+(j%8)
        f.u[r]     = *(const unsigned*)(row + b0 + 2 * r);
        f.u[r + 4] = *(const unsigned*)(row + 16 + b0 + 2 * r);
      }
      afr[i] = f.v;
    }
    #pragma unroll
    for (int j = 0; j < 2; ++j) {
      union { unsigned u[8]; v16bf v; } f;
      const unsigned short* col = Bs[buf][waveN * 32 + j * 16 + lm];
      #pragma unroll
      for (int r = 0; r < 8; ++r)                     // lane: n=lm, K(j)=16*(j/8)+8*half+(j%8)
        f.u[r] = *(const unsigned*)(col + 8 * half + 2 * r + 8 * (r >> 2));
      bfr[j] = f.v;
    }

    // ---- overlap: issue next chunk's copies into the other buffer ----
    if (k0 + 32 < K) stage(k0 + 32, buf ^ 1);

    #pragma unroll
    for (int i = 0; i < 2; ++i)
      #pragma unroll
      for (int j = 0; j < 2; ++j)
        acc[i][j] = __builtin_amdgcn_wmma_f32_16x16x32_bf16(
            false, afr[i], false, bfr[j], (short)0, acc[i][j], false, false);

    wait_async0();
    __syncthreads();
  }

  // ---- epilogue; C/D layout: VGPR r -> M = 8*half + r, N = lm ----
  #pragma unroll
  for (int i = 0; i < 2; ++i) {
    #pragma unroll
    for (int j = 0; j < 2; ++j) {
      int col = nBlk + waveN * 32 + j * 16 + lm;
      float bv = bias[col];
      #pragma unroll
      for (int r = 0; r < 8; ++r) {
        int row = mBlk + waveM * 32 + i * 16 + half * 8 + r;
        if (row < M) {
          float v = acc[i][j][r] + bv;
          size_t o = (size_t)row * Nc + col;
          if constexpr (EPI == 0) {
            outF[o] = v;
          } else if constexpr (EPI == 1) {
            float g = 0.5f * v * (1.0f + erff(v * 0.70710678118654752f));
            outB[o] = f32_to_bf16(g);
          } else {
            float sn, cs;
            __sincosf(v, &sn, &cs);
            outC[o] = cs;
            outS[o] = sn;
          }
        }
      }
    }
  }
}

// ---------------------------------------------------------------------------
// bundle rotation: h(b,0,t)=c*x(b,0,t)-s*x(b,1,t); h(b,1,t)=s*x(b,0,t)+c*x(b,1,t)
// channels per bundle are exactly one float4: (i=0,t=0),(0,1),(1,0),(1,1)
// ---------------------------------------------------------------------------
__global__ void rotate_fwd(const float* __restrict__ x, const float* __restrict__ cosv,
                           const float* __restrict__ sinv, float* __restrict__ h,
                           float* __restrict__ cur, int NB) {
  int i = blockIdx.x * 256 + threadIdx.x;
  if (i >= NB) return;
  float c = cosv[i], s = sinv[i];
  float4 xv = ((const float4*)x)[i];
  float4 o;
  o.x = c * xv.x - s * xv.z;
  o.y = c * xv.y - s * xv.w;
  o.z = s * xv.x + c * xv.z;
  o.w = s * xv.y + c * xv.w;
  ((float4*)h)[i]   = o;
  ((float4*)cur)[i] = o;
}

// ---------------------------------------------------------------------------
// scatter: agg[dst[e]] += ew[e] * cur[src[e]]   (L2-resident f32 atomics)
// thread = (edge, float4-column)
// ---------------------------------------------------------------------------
__global__ void scatter_k(const long long* __restrict__ src, const long long* __restrict__ dst,
                          const float* __restrict__ ew, const float* __restrict__ cur,
                          float* __restrict__ agg, int E) {
  int idx = blockIdx.x * 256 + threadIdx.x;
  int e = idx >> 7, c4 = idx & 127;
  if (e >= E) return;
  int s = (int)src[e], d = (int)dst[e];
  float w = ew[e];
  float4 v = ((const float4*)cur)[(size_t)s * B_DIM + c4];
  float* out = agg + (size_t)d * C_DIM + c4 * 4;
  unsafeAtomicAdd(out + 0, w * v.x);
  unsafeAtomicAdd(out + 1, w * v.y);
  unsafeAtomicAdd(out + 2, w * v.z);
  unsafeAtomicAdd(out + 3, w * v.w);
}

// cur = coef*(cur - agg); h += cur;  optionally emit h as bf16 (final round)
__global__ void diffuse_update(float* __restrict__ cur, const float* __restrict__ agg,
                               float* __restrict__ h, unsigned short* __restrict__ hbf,
                               float coef, int n4) {
  int i = blockIdx.x * 256 + threadIdx.x;
  if (i >= n4) return;
  float4 c = ((const float4*)cur)[i];
  float4 a = ((const float4*)agg)[i];
  float4 nc = make_float4(coef * (c.x - a.x), coef * (c.y - a.y),
                          coef * (c.z - a.z), coef * (c.w - a.w));
  ((float4*)cur)[i] = nc;
  float4 hv = ((const float4*)h)[i];
  hv.x += nc.x; hv.y += nc.y; hv.z += nc.z; hv.w += nc.w;
  ((float4*)h)[i] = hv;
  if (hbf) {   // fused h -> bf16 for GEMM3 input (saves a 41MB pass)
    union { unsigned short s[4]; uint2 u; } p;
    p.s[0] = f32_to_bf16(hv.x); p.s[1] = f32_to_bf16(hv.y);
    p.s[2] = f32_to_bf16(hv.z); p.s[3] = f32_to_bf16(hv.w);
    ((uint2*)hbf)[i] = p.u;
  }
}

// ---------------------------------------------------------------------------
// rotate back with R^T + residual
// ---------------------------------------------------------------------------
__global__ void rotate_bwd(const float* __restrict__ x, const float* __restrict__ hlin,
                           const float* __restrict__ cosv, const float* __restrict__ sinv,
                           float* __restrict__ out, int NB) {
  int i = blockIdx.x * 256 + threadIdx.x;
  if (i >= NB) return;
  float c = cosv[i], s = sinv[i];
  float4 hv = ((const float4*)hlin)[i];
  float4 xv = ((const float4*)x)[i];
  float4 o;
  o.x = xv.x + c * hv.x + s * hv.z;
  o.y = xv.y + c * hv.y + s * hv.w;
  o.z = xv.z - s * hv.x + c * hv.z;
  o.w = xv.w - s * hv.y + c * hv.w;
  ((float4*)out)[i] = o;
}

// ---------------------------------------------------------------------------
// BatchNorm over rows (axis 0): partial column sums + finalize + normalize
// ---------------------------------------------------------------------------
__global__ void bn_stats(const float* __restrict__ h, float* __restrict__ sum,
                         float* __restrict__ sq, int N) {
  int col = blockIdx.x * 128 + threadIdx.x;   // gridDim.x = C/128
  float s = 0.f, q = 0.f;
  for (int r = blockIdx.y; r < N; r += gridDim.y) {
    float v = h[(size_t)r * C_DIM + col];
    s += v; q += v * v;
  }
  unsafeAtomicAdd(&sum[col], s);
  unsafeAtomicAdd(&sq[col], q);
}

__global__ void bn_finalize(const float* __restrict__ sum, const float* __restrict__ sq,
                            const float* __restrict__ gamma, const float* __restrict__ beta,
                            float* __restrict__ scale, float* __restrict__ shift, int N) {
  int c = blockIdx.x * 128 + threadIdx.x;
  if (c >= C_DIM) return;
  float inv = 1.0f / (float)N;
  float mean = sum[c] * inv;
  float var  = sq[c] * inv - mean * mean;
  float sc   = gamma[c] * rsqrtf(var + 1e-5f);
  scale[c] = sc;
  shift[c] = beta[c] - mean * sc;
}

__global__ void bn_normalize(const float* __restrict__ h, const float* __restrict__ scale,
                             const float* __restrict__ shift, float* __restrict__ out, int n4) {
  int i = blockIdx.x * 256 + threadIdx.x;
  if (i >= n4) return;
  int c4 = i & (B_DIM - 1);
  float4 v  = ((const float4*)h)[i];
  float4 sc = ((const float4*)scale)[c4];
  float4 sh = ((const float4*)shift)[c4];
  ((float4*)out)[i] = make_float4(v.x * sc.x + sh.x, v.y * sc.y + sh.y,
                                  v.z * sc.z + sh.z, v.w * sc.w + sh.w);
}

// ---------------------------------------------------------------------------
extern "C" void kernel_launch(void* const* d_in, const int* in_sizes, int n_in,
                              void* d_out, int out_size, void* d_ws, size_t ws_size,
                              hipStream_t stream) {
  const float* x     = (const float*)d_in[0];
  const float* W1    = (const float*)d_in[1];
  const float* b1    = (const float*)d_in[2];
  const float* W2    = (const float*)d_in[3];
  const float* b2    = (const float*)d_in[4];
  const float* Wlin  = (const float*)d_in[5];
  const float* blin  = (const float*)d_in[6];
  const float* gamma = (const float*)d_in[7];
  const float* beta  = (const float*)d_in[8];
  const long long* ei = (const long long*)d_in[9];   // int64 edge_index [2, E]

  const int N = in_sizes[0] / C_DIM;
  const int E = in_sizes[9] / 2;
  const long long* src = ei;
  const long long* dst = ei + E;

  // workspace carve-out (256B aligned)
  char* ws = (char*)d_ws;
  size_t off = 0;
  auto alloc = [&](size_t bytes) -> void* {
    void* p = ws + off;
    off += (bytes + 255) & ~(size_t)255;
    return p;
  };

  float*          deg  = (float*)alloc((size_t)N * 4);
  float*          ew   = (float*)alloc((size_t)E * 4);
  unsigned short* xb   = (unsigned short*)alloc((size_t)N * C_DIM * 2);
  unsigned short* w1b  = (unsigned short*)alloc((size_t)C_DIM * C_DIM * 2);
  unsigned short* w2b  = (unsigned short*)alloc((size_t)C_DIM * B_DIM * 2);
  unsigned short* wlb  = (unsigned short*)alloc((size_t)C_DIM * C_DIM * 2);
  unsigned short* y1b  = (unsigned short*)alloc((size_t)N * C_DIM * 2);
  float*          h    = (float*)alloc((size_t)N * C_DIM * 4);
  float*          cur  = (float*)alloc((size_t)N * C_DIM * 4);
  float*          agg  = (float*)alloc((size_t)N * C_DIM * 4);
  float*          csum = (float*)alloc((size_t)C_DIM * 4);
  float*          csq  = (float*)alloc((size_t)C_DIM * 4);
  float*          scl  = (float*)alloc((size_t)C_DIM * 4);
  float*          shf  = (float*)alloc((size_t)C_DIM * 4);

  // lifetime-based aliases (xb free after GEMM1; y1b free after GEMM2;
  // agg/cur free after the diffusion loop)
  float*          cosv = (float*)xb;                        // N*128 f32
  float*          sinv = (float*)xb + (size_t)N * B_DIM;    // N*128 f32
  unsigned short* hb   = y1b;                               // N*512 bf16
  float*          hlin = agg;                               // N*512 f32
  float*          hfin = cur;                               // N*512 f32

  const int nx = N * C_DIM;
  const int NB = N * B_DIM;

  // --- edge weights ---
  hipMemsetAsync(deg, 0, (size_t)N * 4, stream);
  deg_kernel<<<(E + 255) / 256, 256, 0, stream>>>(src, deg, E);
  ew_kernel<<<(E + 255) / 256, 256, 0, stream>>>(dst, deg, ew, E);

  // --- bf16 conversions for GEMM operands ---
  cvt_bf16_k<<<(nx + 255) / 256, 256, 0, stream>>>(x, xb, nx);
  cvt_bf16_k<<<(C_DIM * C_DIM + 255) / 256, 256, 0, stream>>>(W1, w1b, C_DIM * C_DIM);
  cvt_bf16_k<<<(C_DIM * B_DIM + 255) / 256, 256, 0, stream>>>(W2, w2b, C_DIM * B_DIM);
  cvt_bf16_k<<<(C_DIM * C_DIM + 255) / 256, 256, 0, stream>>>(Wlin, wlb, C_DIM * C_DIM);

  dim3 blk(256);
  // --- struct encoder: GEMM1 (gelu -> bf16), GEMM2 (bias -> cos/sin) ---
  dim3 g1(C_DIM / 64, (N + 127) / 128);
  gemm_bf16_wmma<1><<<g1, blk, 0, stream>>>(xb, w1b, b1, nullptr, y1b, nullptr, nullptr,
                                            N, C_DIM, C_DIM);
  dim3 g2(B_DIM / 64, (N + 127) / 128);
  gemm_bf16_wmma<2><<<g2, blk, 0, stream>>>(y1b, w2b, b2, nullptr, nullptr, cosv, sinv,
                                            N, B_DIM, C_DIM);

  // --- rotate into bundle frame; init diffusion state ---
  rotate_fwd<<<(NB + 255) / 256, 256, 0, stream>>>(x, cosv, sinv, h, cur, NB);

  // --- truncated heat-kernel diffusion (scatter, L2-resident) ---
  for (int k = 1; k <= 4; ++k) {
    hipMemsetAsync(agg, 0, (size_t)N * C_DIM * 4, stream);
    int tot = E * B_DIM;
    scatter_k<<<(tot + 255) / 256, 256, 0, stream>>>(src, dst, ew, cur, agg, E);
    float coef = -1.0f / (float)k;   // TAU = 1
    diffuse_update<<<(NB + 255) / 256, 256, 0, stream>>>(
        cur, agg, h, (k == 4) ? hb : (unsigned short*)nullptr, coef, NB);
  }

  // --- GEMM3: h @ Wlin + blin (hb produced by final diffuse_update) ---
  dim3 g3(C_DIM / 64, (N + 127) / 128);
  gemm_bf16_wmma<0><<<g3, blk, 0, stream>>>(hb, wlb, blin, hlin, nullptr, nullptr, nullptr,
                                            N, C_DIM, C_DIM);

  // --- rotate back (R^T) + residual ---
  rotate_bwd<<<(NB + 255) / 256, 256, 0, stream>>>(x, hlin, cosv, sinv, hfin, NB);

  // --- BatchNorm1d (training-mode, biased var) ---
  hipMemsetAsync(csum, 0, (size_t)C_DIM * 4, stream);
  hipMemsetAsync(csq, 0, (size_t)C_DIM * 4, stream);
  dim3 gs(C_DIM / 128, 256);
  bn_stats<<<gs, 128, 0, stream>>>(hfin, csum, csq, N);
  bn_finalize<<<(C_DIM + 127) / 128, 128, 0, stream>>>(csum, csq, gamma, beta, scl, shf, N);
  bn_normalize<<<(NB + 255) / 256, 256, 0, stream>>>(hfin, scl, shf, (float*)d_out, NB);
}
